// MoELayer_45071386804279
// MI455X (gfx1250) — matrix operand order repaired and checked by reading, not compiled, not run on hardware
//
#include <hip/hip_runtime.h>
#include <hip/hip_bf16.h>
#include <math.h>

// ---- problem constants (match reference) ----
static constexpr int BB = 4, SS = 2048, HH = 1024, FF = 4096, EE = 8;
static constexpr int TT = BB * SS;              // 8192 tokens
static constexpr int TM = 16;                   // tokens per tile (WMMA M)
static constexpr int FC = 128;                  // F-chunk width
static constexpr int XS = HH + 8;               // LDS row stride (halves) for X tile
static constexpr int HSD = FC + 8;              // LDS row stride (halves) for H chunk
static constexpr int FRAG = 512;                // halves per 32x16 B fragment (1 KB)

typedef __attribute__((ext_vector_type(16))) __bf16 v16bf;
typedef __attribute__((ext_vector_type(8)))  float  v8f;

__device__ __forceinline__ unsigned short f2bf(float f) {
    unsigned u = __float_as_uint(f);
    unsigned r = (u + 0x7FFFu + ((u >> 16) & 1u)) >> 16;   // round-to-nearest-even
    return (unsigned short)r;
}

__device__ __forceinline__ v8f zero8() {
    v8f z = {0.f,0.f,0.f,0.f,0.f,0.f,0.f,0.f};
    return z;
}

// A fragment (16xK bf16, M=row) from LDS, ISA 7.12.2 16-bit A layout:
//  lanes 0-15: row=lane,   halves[0..7]=K k0..k0+7,  halves[8..15]=K k0+16..k0+23
//  lanes16-31: row=lane-16, K offsets +8 / +24
__device__ __forceinline__ v16bf load_afrag_lds(const unsigned short* base, int stride,
                                                int lane, int k0) {
    int row = lane & 15;
    int khi = (lane >> 4) ? 8 : 0;
    const unsigned short* p = base + row * stride + k0 + khi;
    union { v16bf v; uint4 q[2]; } u;
    u.q[0] = *(const uint4*)(p);        // 8 halves
    u.q[1] = *(const uint4*)(p + 16);   // 8 halves, K += 16
    return u.v;
}

// B fragment from pre-swizzled bf16 weights: frag is 1 KB, [lane][16 halves].
// Two fully-coalesced global_load_b128 per wave, no VALU.
__device__ __forceinline__ v16bf load_bfrag_t(const unsigned short* __restrict__ frag,
                                              int lane) {
    const uint4* p = (const uint4*)frag + lane * 2;
    union { v16bf v; uint4 q[2]; } u;
    u.q[0] = p[0];
    u.q[1] = p[1];
    return u.v;
}

// ------- weight pre-pass: fp32 [E][Kd][Nd] -> bf16 B-fragments [e][nt][kt][lane][16] ----
__global__ __launch_bounds__(256)
void convert_w_kernel(const float* __restrict__ src, unsigned short* __restrict__ dst,
                      int Kd, int Nd) {
    int wid  = (blockIdx.x * blockDim.x + threadIdx.x) >> 5;   // one wave per fragment
    int lane = threadIdx.x & 31;
    int ktiles = Kd >> 5, ntiles = Nd >> 4;
    int fpe = ktiles * ntiles;
    if (wid >= EE * fpe) return;
    int e  = wid / fpe, rem = wid - e * fpe;
    int nt = rem / ktiles, kt = rem - nt * ktiles;             // layout [e][nt][kt]
    const float* s = src + (size_t)e * Kd * Nd;
    int k0 = kt * 32 + ((lane >> 4) ? 8 : 0);                  // B layout: hi lanes K+8, halves 8..15 K+16
    int n  = nt * 16 + (lane & 15);
    union { uint4 q[2]; unsigned u[8]; } pk;
#pragma unroll
    for (int i = 0; i < 8; ++i) {
        unsigned lo = f2bf(s[(size_t)(k0 + 2 * i + (i >= 4 ? 8 : 0))     * Nd + n]);
        unsigned hi = f2bf(s[(size_t)(k0 + 2 * i + 1 + (i >= 4 ? 8 : 0)) * Nd + n]);
        pk.u[i] = lo | (hi << 16);
    }
    uint4* d = (uint4*)(dst + (size_t)wid * FRAG) + lane * 2;
    d[0] = pk.q[0];
    d[1] = pk.q[1];
}

// ---------------- gating: softmax -> top-2 -> renorm -> compact per-expert lists ----
__global__ __launch_bounds__(256)
void moe_gate_kernel(const float* __restrict__ x, const float* __restrict__ gate_w,
                     const float* __restrict__ gate_b,
                     int* __restrict__ counts, int* __restrict__ idx,
                     float* __restrict__ wts) {
    int tok  = (blockIdx.x * blockDim.x + threadIdx.x) >> 5;   // one wave per token
    int lane = threadIdx.x & 31;
    if (tok >= TT) return;

    const float* xr = x + (size_t)tok * HH;
    float acc[EE];
#pragma unroll
    for (int e = 0; e < EE; ++e) acc[e] = 0.f;

    for (int k = lane; k < HH; k += 32) {
        float xv = xr[k];
        const float4* gw = (const float4*)(gate_w + (size_t)k * EE);
        float4 g0 = gw[0], g1 = gw[1];
        acc[0] += xv * g0.x; acc[1] += xv * g0.y;
        acc[2] += xv * g0.z; acc[3] += xv * g0.w;
        acc[4] += xv * g1.x; acc[5] += xv * g1.y;
        acc[6] += xv * g1.z; acc[7] += xv * g1.w;
    }
#pragma unroll
    for (int off = 16; off >= 1; off >>= 1) {
#pragma unroll
        for (int e = 0; e < EE; ++e) acc[e] += __shfl_xor(acc[e], off, 32);
    }
    if (lane == 0) {
        float z[EE], m = -1e30f;
#pragma unroll
        for (int e = 0; e < EE; ++e) { z[e] = acc[e] + gate_b[e]; m = fmaxf(m, z[e]); }
        float p[EE];
#pragma unroll
        for (int e = 0; e < EE; ++e) p[e] = __expf(z[e] - m);
        int i1 = 0;
#pragma unroll
        for (int e = 1; e < EE; ++e) if (p[e] > p[i1]) i1 = e;
        int i2 = (i1 == 0) ? 1 : 0;
#pragma unroll
        for (int e = 0; e < EE; ++e) if (e != i1 && p[e] > p[i2]) i2 = e;
        float inv = 1.f / (p[i1] + p[i2]);      // softmax denom cancels in renorm
        int p1 = atomicAdd(&counts[i1], 1);
        idx[i1 * TT + p1] = tok;  wts[i1 * TT + p1] = p[i1] * inv;
        int p2 = atomicAdd(&counts[i2], 1);
        idx[i2 * TT + p2] = tok;  wts[i2 * TT + p2] = p[i2] * inv;
    }
}

// ---------------- fused routed FFN: y += relu(x@W1+b1)@W2 + b2, scaled by combine ----
__global__ __launch_bounds__(256)
void moe_ffn_kernel(const float* __restrict__ x,
                    const unsigned short* __restrict__ w1t,
                    const float* __restrict__ b1,
                    const unsigned short* __restrict__ w2t,
                    const float* __restrict__ b2,
                    const int* __restrict__ counts, const int* __restrict__ idx,
                    const float* __restrict__ wts, float* __restrict__ out) {
    __shared__ __align__(16) unsigned short Xs[TM * XS];   // 16x1024 bf16 x-tile
    __shared__ __align__(16) unsigned short Hs[TM * HSD];  // 16x128  bf16 h-chunk
    __shared__ int   tok_s[TM];
    __shared__ float wt_s[TM];

    const int e  = blockIdx.y;
    const int mt = blockIdx.x;
    const int cnt = counts[e];
    if (mt * TM >= cnt) return;                 // uniform per block: EXEC stays full

    const int tid = threadIdx.x, lane = tid & 31, wv = tid >> 5;   // 8 waves

    if (tid < TM) {
        int r = mt * TM + tid;
        if (r < cnt) { tok_s[tid] = idx[e * TT + r]; wt_s[tid] = wts[e * TT + r]; }
        else         { tok_s[tid] = 0;               wt_s[tid] = 0.f; }
    }
    __syncthreads();

    // gather + downconvert the 16-token x tile into LDS
#pragma unroll
    for (int rr = 0; rr < 2; ++rr) {
        int row = wv * 2 + rr;
        const float4* xr = (const float4*)(x + (size_t)tok_s[row] * HH);
        for (int c4 = lane; c4 < HH / 4; c4 += 32) {
            float4 v = xr[c4];
            unsigned v01 = (unsigned)f2bf(v.x) | ((unsigned)f2bf(v.y) << 16);
            unsigned v23 = (unsigned)f2bf(v.z) | ((unsigned)f2bf(v.w) << 16);
            *(uint2*)&Xs[row * XS + c4 * 4] = make_uint2(v01, v23);
        }
    }
    __syncthreads();

    // pre-swizzled weights for this expert, layout [ntile][ktile] x 1KB fragments
    const unsigned short* w1te = w1t + (size_t)e * HH * FF;    // (HH/32 * FF/16 frags)
    const unsigned short* w2te = w2t + (size_t)e * FF * HH;

    v8f yacc[8];                               // wave owns out cols [wv*128, wv*128+128)
#pragma unroll
    for (int j = 0; j < 8; ++j) yacc[j] = zero8();

    const int rbase = (lane >> 4) * 8;         // C/D layout: rows rbase..rbase+7
    const int nlane = lane & 15;

    for (int fc = 0; fc < FF; fc += FC) {
        // ---- stage 1: h tile = relu(X @ W1[:, fcol0:fcol0+16] + b1) ----
        const int fcol0 = fc + wv * 16;
        const unsigned short* fb = w1te + (size_t)(fcol0 >> 4) * (HH >> 5) * FRAG;
        if (fc + FC < FF)                       // next chunk's first fragment
            __builtin_prefetch(w1te + (size_t)((fcol0 + FC) >> 4) * (HH >> 5) * FRAG, 0, 1);
        v8f hacc = zero8();
#pragma unroll 4
        for (int k0 = 0; k0 < HH; k0 += 32) {
            v16bf a = load_afrag_lds(Xs, XS, lane, k0);
            v16bf b = load_bfrag_t(fb, lane);
            fb += FRAG;                         // [nt][kt]: K-walk is contiguous 1KB steps
            hacc = __builtin_amdgcn_wmma_f32_16x16x32_bf16(
                false, a, false, b, (short)0, hacc, false, false);
        }
        float b1v = b1[e * FF + fcol0 + nlane];

        __syncthreads();                        // prior chunk's Hs reads complete
        const int hcol = wv * 16 + nlane;       // chunk-relative K position
#pragma unroll
        for (int r = 0; r < 8; ++r) {
            float hv = fmaxf(hacc[r] + b1v, 0.f);
            Hs[(rbase + r) * HSD + hcol] = f2bf(hv);
        }
        __syncthreads();

        // ---- stage 2: yacc += Hchunk @ W2[fc:fc+128, wv*128 : wv*128+128] ----
#pragma unroll
        for (int kc = 0; kc < FC; kc += 32) {
            v16bf a = load_afrag_lds(Hs, HSD, lane, kc);
            const int ktf = (fc + kc) >> 5;
#pragma unroll
            for (int j = 0; j < 8; ++j) {
                const int ntf = wv * 8 + j;     // (wv*128 + j*16) / 16
                const unsigned short* fb2 =
                    w2te + ((size_t)ntf * (FF >> 5) + ktf) * FRAG;
                v16bf b = load_bfrag_t(fb2, lane);
                yacc[j] = __builtin_amdgcn_wmma_f32_16x16x32_bf16(
                    false, a, false, b, (short)0, yacc[j], false, false);
            }
        }
    }

    // ---- epilogue: +b2, scale by combine weight, scatter-add into out ----
#pragma unroll
    for (int j = 0; j < 8; ++j) {
        int col = wv * 128 + j * 16 + nlane;
        float bias = b2[e * HH + col];
#pragma unroll
        for (int r = 0; r < 8; ++r) {
            int row = rbase + r;
            float v = (yacc[j][r] + bias) * wt_s[row];
            atomicAdd(&out[(size_t)tok_s[row] * HH + col], v);
        }
    }
}

// ---------------------------------------------------------------------------------
extern "C" void kernel_launch(void* const* d_in, const int* in_sizes, int n_in,
                              void* d_out, int out_size, void* d_ws, size_t ws_size,
                              hipStream_t stream) {
    (void)in_sizes; (void)n_in; (void)out_size; (void)ws_size;
    const float* x      = (const float*)d_in[0];
    const float* gate_w = (const float*)d_in[1];
    const float* gate_b = (const float*)d_in[2];
    const float* w1     = (const float*)d_in[3];
    const float* b1     = (const float*)d_in[4];
    const float* w2     = (const float*)d_in[5];
    const float* b2     = (const float*)d_in[6];
    float* out = (float*)d_out;

    // workspace: [counts 16 ints][idx E*T ints][wts E*T f32][w1t bf16][w2t bf16]
    int*   counts = (int*)d_ws;
    int*   idx    = counts + 16;
    float* wts    = (float*)(idx + EE * TT);
    unsigned short* w1t = (unsigned short*)(wts + EE * TT);        // E*H*F bf16 (64 MB)
    unsigned short* w2t = w1t + (size_t)EE * HH * FF;              // E*F*H bf16 (64 MB)

    hipMemsetAsync(counts, 0, 16 * sizeof(int), stream);
    hipMemsetAsync(out, 0, (size_t)TT * HH * sizeof(float), stream);

    // weight downconvert + swizzle into WMMA-B fragment layout
    {
        int waves = EE * (HH >> 5) * (FF >> 4);                    // 65536 fragments
        convert_w_kernel<<<dim3(waves / 8), dim3(256), 0, stream>>>(w1, w1t, HH, FF);
        waves = EE * (FF >> 5) * (HH >> 4);
        convert_w_kernel<<<dim3(waves / 8), dim3(256), 0, stream>>>(w2, w2t, FF, HH);
    }

    moe_gate_kernel<<<dim3((TT * 32) / 256), dim3(256), 0, stream>>>(
        x, gate_w, gate_b, counts, idx, wts);

    moe_ffn_kernel<<<dim3(TT / TM, EE), dim3(256), 0, stream>>>(
        x, w1t, b1, w2t, b2, counts, idx, wts, out);
}
